// HFBlock_73538430042690
// MI455X (gfx1250) — matrix operand order repaired and checked by reading, not compiled, not run on hardware
//
#include <hip/hip_runtime.h>
#include <hip/hip_bf16.h>
#include <math.h>

#define BN 4
#define DM 128
#define HH 64
#define WW 64
#define LL 4096            // HH*WW
#define DI 256
#define DS 16
#define DTR 8
#define KD 4
#define NL 2
#define CPROJ 40           // DT_RANK + 2*D_STATE
#define EPSV 1e-5f

typedef __attribute__((ext_vector_type(16))) _Float16 v16h;
typedef __attribute__((ext_vector_type(8)))  float    v8f;

__device__ __forceinline__ float sigmoidf_(float x) { return 1.f / (1.f + __expf(-x)); }
__device__ __forceinline__ float siluf_(float x)    { return x * sigmoidf_(x); }
__device__ __forceinline__ float softplusf_(float x){ return (x > 20.f) ? x : log1pf(__expf(x)); }

// ---------------------------------------------------------------------------
// WMMA helpers: one wave32 computes a 16x16 f32 tile, K stepped by 32 (f16).
// Fragment layouts per CDNA5 ISA 7.12.2:
//   A (16x32, 16-bit): lane m = lane&15, half = lane>>4;
//     elem j -> VGPR v=j>>1, k = 2*(v&3) + 8*half + (v>=4 ? 16 : 0) + (j&1)
//   B (32x16, 16-bit): lane n = lane&15, half = lane>>4; elem j -> same k as A
//   C/D (16x16 f32):   VGPR r -> M = r + 8*half, N = lane&15
// ---------------------------------------------------------------------------
template <typename LA, typename LB>
__device__ __forceinline__ void wmma_k32(v8f& acc, int lane, int k0, LA loadA, LB loadB) {
    const int m    = lane & 15;
    const int n    = lane & 15;
    const int half = lane >> 4;
    v16h a, b;
#pragma unroll
    for (int j = 0; j < 16; ++j) {
        const int v  = j >> 1;
        const int ka = ((v & 3) << 1) + (half << 3) + ((v >= 4) ? 16 : 0) + (j & 1);
        a[j] = (_Float16)loadA(m, k0 + ka);
        b[j] = (_Float16)loadB(k0 + ka, n);
    }
    acc = __builtin_amdgcn_wmma_f32_16x16x32_f16(false, a, false, b, (short)0, acc, false, false);
}

template <typename ST>
__device__ __forceinline__ void wmma_store16x16(const v8f& acc, int lane, ST store) {
    const int n    = lane & 15;
    const int half = lane >> 4;
#pragma unroll
    for (int r = 0; r < 8; ++r) store(r + (half << 3), n, acc[r]);
}

// ---------------------------------------------------------------------------
// 1) in_proj: xz[b,e,l] = sum_c inw[e,c] * x[b,c,l]   (M=512,K=128,N=4096/b)
// grid (32, 256, B), block 32
// ---------------------------------------------------------------------------
__global__ void k_inproj(const float* __restrict__ inw, const float* __restrict__ x,
                         float* __restrict__ xz) {
    const int lane = threadIdx.x & 31;
    const int e0 = blockIdx.x << 4;
    const int l0 = blockIdx.y << 4;
    const int b  = blockIdx.z;
    v8f acc = {};
    auto lA = [&](int m, int k) { return inw[(e0 + m) * DM + k]; };
    auto lB = [&](int k, int n) { return x[((size_t)b * DM + k) * LL + l0 + n]; };
#pragma unroll
    for (int k0 = 0; k0 < DM; k0 += 32) wmma_k32(acc, lane, k0, lA, lB);
    wmma_store16x16(acc, lane, [&](int m, int n, float v) {
        xz[((size_t)b * (2 * DI) + e0 + m) * LL + l0 + n] = v;
    });
}

// ---------------------------------------------------------------------------
// 2) depthwise 3x3 conv + bias + SiLU on xz[:, :DI]  -> xi[b,d,l]
// ---------------------------------------------------------------------------
__global__ void k_dwconv_silu(const float* __restrict__ xz, const float* __restrict__ cw,
                              const float* __restrict__ cb, float* __restrict__ xi) {
    const size_t i = (size_t)blockIdx.x * blockDim.x + threadIdx.x; // (b,d,l)
    const int l = i & (LL - 1);
    const int d = (i >> 12) & (DI - 1);
    const int b = (int)(i >> 20);
    const int h = l >> 6, w = l & 63;
    const float* src = xz + ((size_t)b * (2 * DI) + d) * LL;
    float s = cb[d];
#pragma unroll
    for (int dh = -1; dh <= 1; ++dh)
#pragma unroll
        for (int dw = -1; dw <= 1; ++dw) {
            const int hh = h + dh, ww = w + dw;
            if (hh >= 0 && hh < HH && ww >= 0 && ww < WW)
                s += cw[d * 9 + (dh + 1) * 3 + (dw + 1)] * src[hh * WW + ww];
        }
    xi[i] = siluf_(s);
}

// ---------------------------------------------------------------------------
// 3) build 4 scan orderings: xs[b,k,d,l]
// ---------------------------------------------------------------------------
__global__ void k_build_xs(const float* __restrict__ xi, float* __restrict__ xs) {
    const size_t i = (size_t)blockIdx.x * blockDim.x + threadIdx.x; // (b,d,l)
    const int l = i & (LL - 1);
    const int d = (i >> 12) & (DI - 1);
    const int b = (int)(i >> 20);
    const int h = l >> 6, w = l & 63;
    const float v = xi[i];
    const int lt = w * HH + h;
    float* base = xs + (size_t)b * KD * DI * LL + (size_t)d * LL;
    const size_t kstride = (size_t)DI * LL;
    base[0 * kstride + l]            = v;
    base[1 * kstride + lt]           = v;
    base[2 * kstride + (LL - 1 - l)] = v;
    base[3 * kstride + (LL - 1 - lt)] = v;
}

// ---------------------------------------------------------------------------
// 4) x_proj: xdbl_t[b,k,l,c] = sum_d xpw[k,c,d]*xs[b,k,d,l]  (stored transposed)
// grid (3, 256, B*K), block 32.  M=40 padded to 48.
// ---------------------------------------------------------------------------
__global__ void k_xproj(const float* __restrict__ xpw, const float* __restrict__ xs,
                        float* __restrict__ xdbl_t) {
    const int lane = threadIdx.x & 31;
    const int c0 = blockIdx.x << 4;
    const int l0 = blockIdx.y << 4;
    const int bk = blockIdx.z;               // b*KD + k
    const int kdir = bk & 3;
    v8f acc = {};
    auto lA = [&](int m, int k) {
        const int c = c0 + m;
        return (c < CPROJ) ? xpw[((size_t)kdir * CPROJ + c) * DI + k] : 0.f;
    };
    auto lB = [&](int k, int n) { return xs[((size_t)bk * DI + k) * LL + l0 + n]; };
#pragma unroll
    for (int k0 = 0; k0 < DI; k0 += 32) wmma_k32(acc, lane, k0, lA, lB);
    wmma_store16x16(acc, lane, [&](int m, int n, float v) {
        if (c0 + m < CPROJ)
            xdbl_t[((size_t)bk * LL + l0 + n) * CPROJ + c0 + m] = v;
    });
}

// ---------------------------------------------------------------------------
// 5) dt projection + softplus: dts[b,k,d,l]
// ---------------------------------------------------------------------------
__global__ void k_dtproj(const float* __restrict__ xdbl_t, const float* __restrict__ dtw,
                         const float* __restrict__ dtb, float* __restrict__ dts) {
    const size_t i = (size_t)blockIdx.x * blockDim.x + threadIdx.x; // (b,k,d,l)
    const int l = i & (LL - 1);
    const size_t rest = i >> 12;
    const int d  = rest & (DI - 1);
    const int bk = (int)(rest >> 8);
    const int kdir = bk & 3;
    const float* xr = xdbl_t + ((size_t)bk * LL + l) * CPROJ;
    const float* wr = dtw + ((size_t)kdir * DI + d) * DTR;
    float s = dtb[kdir * DI + d];
#pragma unroll
    for (int r = 0; r < DTR; ++r) s += wr[r] * xr[r];
    dts[i] = softplusf_(s);
}

// ---------------------------------------------------------------------------
// 6) selective scan. 4096 chains (b,k,d); lane = state n; 2 chains per wave32.
// grid 256, block 256 (8 waves)
// ---------------------------------------------------------------------------
__global__ void k_scan(const float* __restrict__ xs, const float* __restrict__ dts,
                       const float* __restrict__ xdbl_t, const float* __restrict__ Alog,
                       float* __restrict__ ys) {
    const int lane = threadIdx.x & 31;
    const int wave = blockIdx.x * (blockDim.x >> 5) + (threadIdx.x >> 5);
    const int chain = (wave << 1) + (lane >> 4);   // 0..4095  == bk*256 + d
    const int n = lane & 15;
    const int d  = chain & (DI - 1);
    const int bk = chain >> 8;
    const int kdir = bk & 3;
    const float a = -__expf(Alog[((size_t)kdir * DI + d) * DS + n]);
    const float* xs_c  = xs  + ((size_t)bk * DI + d) * LL;
    const float* dts_c = dts + ((size_t)bk * DI + d) * LL;
    const float* xd    = xdbl_t + (size_t)bk * LL * CPROJ;
    float* ys_c        = ys  + ((size_t)bk * DI + d) * LL;
    float h = 0.f;
    for (int l = 0; l < LL; ++l) {
        if ((l & 63) == 0 && l + 64 < LL)
            __builtin_prefetch(xd + (size_t)(l + 64) * CPROJ, 0, 1);
        const float u  = xs_c[l];
        const float dt = dts_c[l];
        const float Bt = xd[(size_t)l * CPROJ + DTR + n];
        const float Ct = xd[(size_t)l * CPROJ + DTR + DS + n];
        h = __expf(dt * a) * h + (dt * u) * Bt;
        float y = h * Ct;
        y += __shfl_xor(y, 1, 32);
        y += __shfl_xor(y, 2, 32);
        y += __shfl_xor(y, 4, 32);
        y += __shfl_xor(y, 8, 32);
        if (n == 0) ys_c[l] = y;
    }
}

// ---------------------------------------------------------------------------
// 7) merge 4 directions + D skip + LayerNorm(d) + SiLU(z) gate -> yo[b,l,d]
// grid B*L, block 256
// ---------------------------------------------------------------------------
__global__ void k_combine(const float* __restrict__ ys, const float* __restrict__ xs,
                          const float* __restrict__ Dsk, const float* __restrict__ ong,
                          const float* __restrict__ onb, const float* __restrict__ xz,
                          float* __restrict__ yo) {
    const int b = blockIdx.x >> 12;
    const int l = blockIdx.x & (LL - 1);
    const int d = threadIdx.x;
    const int h = l >> 6, w = l & 63;
    const int lt = w * HH + h;
    const size_t base = (size_t)b * KD * DI * LL;
    auto Y = [&](int k, int li) {
        const size_t off = base + ((size_t)k * DI + d) * LL + li;
        return ys[off] + Dsk[k * DI + d] * xs[off];
    };
    const float v = Y(0, l) + Y(1, lt) + Y(2, LL - 1 - l) + Y(3, LL - 1 - lt);

    __shared__ float sbuf[DI];
    sbuf[d] = v;
    __syncthreads();
    for (int s = DI / 2; s > 0; s >>= 1) { if (d < s) sbuf[d] += sbuf[d + s]; __syncthreads(); }
    const float mu = sbuf[0] * (1.f / DI);
    __syncthreads();
    const float dv = v - mu;
    sbuf[d] = dv * dv;
    __syncthreads();
    for (int s = DI / 2; s > 0; s >>= 1) { if (d < s) sbuf[d] += sbuf[d + s]; __syncthreads(); }
    const float var = sbuf[0] * (1.f / DI);
    const float normed = dv * rsqrtf(var + EPSV) * ong[d] + onb[d];
    const float z = xz[((size_t)b * (2 * DI) + DI + d) * LL + l];
    yo[((size_t)b * LL + l) * DI + d] = normed * siluf_(z);
}

// ---------------------------------------------------------------------------
// 8) out_proj + residual: x[b,c,l] += sum_d opw[c,d] * yo[b,l,d]
// grid (8, 256, B), block 32
// ---------------------------------------------------------------------------
__global__ void k_outproj(const float* __restrict__ opw, const float* __restrict__ yo,
                          float* __restrict__ xcur) {
    const int lane = threadIdx.x & 31;
    const int c0 = blockIdx.x << 4;
    const int l0 = blockIdx.y << 4;
    const int b  = blockIdx.z;
    v8f acc = {};
    auto lA = [&](int m, int k) { return opw[(c0 + m) * DI + k]; };
    auto lB = [&](int k, int n) { return yo[((size_t)b * LL + l0 + n) * DI + k]; };
#pragma unroll
    for (int k0 = 0; k0 < DI; k0 += 32) wmma_k32(acc, lane, k0, lA, lB);
    wmma_store16x16(acc, lane, [&](int m, int n, float v) {
        xcur[((size_t)b * DM + c0 + m) * LL + l0 + n] += v;
    });
}

// ---------------------------------------------------------------------------
// 9) final 1x1 conv over concat(x, x_shallow): t[b,l,o] = W[o,:]·cat + bias
// grid (16, 256, B), block 32   (stored transposed for the channel LayerNorm)
// ---------------------------------------------------------------------------
__global__ void k_conv1(const float* __restrict__ w1, const float* __restrict__ b1,
                        const float* __restrict__ xcur, const float* __restrict__ xsh,
                        float* __restrict__ tbuf) {
    const int lane = threadIdx.x & 31;
    const int o0 = blockIdx.x << 4;
    const int l0 = blockIdx.y << 4;
    const int b  = blockIdx.z;
    v8f acc = {};
    auto lA = [&](int m, int k) { return w1[(o0 + m) * (2 * DM) + k]; };
    auto lB = [&](int k, int n) {
        return (k < DM) ? xcur[((size_t)b * DM + k) * LL + l0 + n]
                        : xsh[((size_t)b * DM + (k - DM)) * LL + l0 + n];
    };
#pragma unroll
    for (int k0 = 0; k0 < 2 * DM; k0 += 32) wmma_k32(acc, lane, k0, lA, lB);
    wmma_store16x16(acc, lane, [&](int m, int n, float v) {
        tbuf[((size_t)b * LL + l0 + n) * (2 * DM) + o0 + m] = v + b1[o0 + m];
    });
}

// ---------------------------------------------------------------------------
// 10) final LayerNorm(channel) + pixel-shuffle -> out[b, c/4, 2H, 2W]
// grid B*L, block 256
// ---------------------------------------------------------------------------
__global__ void k_final(const float* __restrict__ tbuf, const float* __restrict__ ng,
                        const float* __restrict__ nb, float* __restrict__ out) {
    const int b = blockIdx.x >> 12;
    const int l = blockIdx.x & (LL - 1);
    const int o = threadIdx.x;
    const float v = tbuf[((size_t)b * LL + l) * (2 * DM) + o];

    __shared__ float sbuf[2 * DM];
    sbuf[o] = v;
    __syncthreads();
    for (int s = DM; s > 0; s >>= 1) { if (o < s) sbuf[o] += sbuf[o + s]; __syncthreads(); }
    const float mu = sbuf[0] * (1.f / (2 * DM));
    __syncthreads();
    const float dv = v - mu;
    sbuf[o] = dv * dv;
    __syncthreads();
    for (int s = DM; s > 0; s >>= 1) { if (o < s) sbuf[o] += sbuf[o + s]; __syncthreads(); }
    const float var = sbuf[0] * (1.f / (2 * DM));
    const float nv = dv * rsqrtf(var + EPSV) * ng[o] + nb[o];

    const int h = l >> 6, w = l & 63;
    const int cq = o >> 2, pi = (o >> 1) & 1, pj = o & 1;
    out[(((size_t)b * 64 + cq) * (2 * HH) + (2 * h + pi)) * (2 * WW) + (2 * w + pj)] = nv;
}

// ---------------------------------------------------------------------------
extern "C" void kernel_launch(void* const* d_in, const int* in_sizes, int n_in,
                              void* d_out, int out_size, void* d_ws, size_t ws_size,
                              hipStream_t stream) {
    const float* x_deep    = (const float*)d_in[0];
    const float* x_shallow = (const float*)d_in[1];
    const float* in_proj_w = (const float*)d_in[2];
    const float* conv_w    = (const float*)d_in[3];
    const float* conv_b    = (const float*)d_in[4];
    const float* x_proj_w  = (const float*)d_in[5];
    const float* dt_w      = (const float*)d_in[6];
    const float* dt_b      = (const float*)d_in[7];
    const float* A_logs    = (const float*)d_in[8];
    const float* Ds        = (const float*)d_in[9];
    const float* out_ng    = (const float*)d_in[10];
    const float* out_nb    = (const float*)d_in[11];
    const float* out_proj_w= (const float*)d_in[12];
    const float* conv1_w   = (const float*)d_in[13];
    const float* conv1_b   = (const float*)d_in[14];
    const float* norm_g    = (const float*)d_in[15];
    const float* norm_b    = (const float*)d_in[16];
    float* out = (float*)d_out;

    // workspace carve-up (floats)
    float* ws = (float*)d_ws;
    const size_t N_X    = (size_t)BN * DM * LL;          // 2,097,152
    const size_t N_XZ   = (size_t)BN * 2 * DI * LL;      // 8,388,608
    const size_t N_XI   = (size_t)BN * DI * LL;          // 4,194,304
    const size_t N_XS   = (size_t)BN * KD * DI * LL;     // 16,777,216
    const size_t N_XDBL = (size_t)BN * KD * LL * CPROJ;  // 2,621,440
    const size_t N_DTS  = N_XS;
    const size_t N_YS   = N_XS;
    const size_t N_YO   = N_XI;
    const size_t N_TB   = (size_t)BN * LL * 2 * DM;
    float* xcur   = ws;                  ws += N_X;
    float* xz     = ws;                  ws += N_XZ;
    float* xi     = ws;                  ws += N_XI;
    float* xs     = ws;                  ws += N_XS;
    float* xdbl_t = ws;                  ws += N_XDBL;
    float* dts    = ws;                  ws += N_DTS;
    float* ys     = ws;                  ws += N_YS;
    float* yo     = ws;                  ws += N_YO;
    float* tbuf   = ws;                  ws += N_TB;
    (void)ws_size; (void)in_sizes; (void)n_in; (void)out_size;

    hipMemcpyAsync(xcur, x_deep, N_X * sizeof(float), hipMemcpyDeviceToDevice, stream);

    for (int layer = 0; layer < NL; ++layer) {
        const float* inw = in_proj_w + (size_t)layer * 2 * DI * DM;
        const float* cw  = conv_w    + (size_t)layer * DI * 9;
        const float* cb  = conv_b    + (size_t)layer * DI;
        const float* xpw = x_proj_w  + (size_t)layer * KD * CPROJ * DI;
        const float* dtw = dt_w      + (size_t)layer * KD * DI * DTR;
        const float* dtb = dt_b      + (size_t)layer * KD * DI;
        const float* Al  = A_logs    + (size_t)layer * KD * DI * DS;
        const float* Dk  = Ds        + (size_t)layer * KD * DI;
        const float* ong = out_ng    + (size_t)layer * DI;
        const float* onb = out_nb    + (size_t)layer * DI;
        const float* opw = out_proj_w+ (size_t)layer * DM * DI;

        k_inproj<<<dim3(32, 256, BN), 32, 0, stream>>>(inw, xcur, xz);
        k_dwconv_silu<<<16384, 256, 0, stream>>>(xz, cw, cb, xi);
        k_build_xs<<<16384, 256, 0, stream>>>(xi, xs);
        k_xproj<<<dim3(3, 256, BN * KD), 32, 0, stream>>>(xpw, xs, xdbl_t);
        k_dtproj<<<65536, 256, 0, stream>>>(xdbl_t, dtw, dtb, dts);
        k_scan<<<256, 256, 0, stream>>>(xs, dts, xdbl_t, Al, ys);
        k_combine<<<BN * LL, DI, 0, stream>>>(ys, xs, Dk, ong, onb, xz, yo);
        k_outproj<<<dim3(8, 256, BN), 32, 0, stream>>>(opw, yo, xcur);
    }

    k_conv1<<<dim3(16, 256, BN), 32, 0, stream>>>(conv1_w, conv1_b, xcur, x_shallow, tbuf);
    k_final<<<BN * LL, 2 * DM, 0, stream>>>(tbuf, norm_g, norm_b, out);
}
